// MHA_30374008717345
// MI455X (gfx1250) — matrix operand order repaired and checked by reading, not compiled, not run on hardware
//
#include <hip/hip_runtime.h>

// ---------------- problem constants ----------------
#define DM  2048      // d_model
#define NH  16        // heads
#define DH  128       // head dim
#define BB  2         // batch
#define SQ  2048      // seq
#define BS  (BB*SQ)   // 4096 rows total

typedef __bf16 bf16;
typedef __attribute__((ext_vector_type(4)))  float f32x4;
typedef __attribute__((ext_vector_type(8)))  float f32x8;
typedef __attribute__((ext_vector_type(8)))  bf16  bf16x8;
typedef __attribute__((ext_vector_type(16))) bf16  bf16x16;

// ---------------- fragment helpers ----------------
static __device__ inline bf16x8 cvt8(const float* __restrict__ p) {
  f32x4 a = *(const f32x4*)p;
  f32x4 b = *(const f32x4*)(p + 4);
  f32x8 w = __builtin_shufflevector(a, b, 0,1,2,3,4,5,6,7);
  return __builtin_convertvector(w, bf16x8);
}
static __device__ inline bf16x16 cat8(bf16x8 lo, bf16x8 hi) {
  return __builtin_shufflevector(lo, hi, 0,1,2,3,4,5,6,7,8,9,10,11,12,13,14,15);
}
// A fragment 16x32 (bf16): lane r=lane&15 holds row r; hh=lane>>4 picks K interleave.
// element j -> k = 16*(j>>3) + 8*hh + (j&7): two contiguous 8-element chunks.
static __device__ inline bf16x16 load_afrag(const float* __restrict__ row, int k, int hh) {
  return cat8(cvt8(row + k + 8*hh), cvt8(row + k + 16 + 8*hh));
}
static __device__ inline bf16x16 load_afrag(const bf16* __restrict__ row, int k, int hh) {
  bf16x8 lo = *(const bf16x8*)(row + k + 8*hh);
  bf16x8 hi = *(const bf16x8*)(row + k + 16 + 8*hh);
  return cat8(lo, hi);
}
// B fragment 32x16 (bf16): lane holds one column (= one source row); k = 16*(j>>3) + 8*hh + (j&7).
static __device__ inline bf16x16 load_bfrag(const float* __restrict__ row, int k, int hh) {
  return cat8(cvt8(row + k + 8*hh), cvt8(row + k + 16 + 8*hh));
}
static __device__ inline bf16x16 load_bfrag(const bf16* __restrict__ row, int k, int hh) {
  bf16x8 lo = *(const bf16x8*)(row + k + 8*hh);
  bf16x8 hi = *(const bf16x8*)(row + k + 16 + 8*hh);
  return cat8(lo, hi);
}
static __device__ inline f32x8 wmma_bf16(bf16x16 a, bf16x16 b, f32x8 c) {
  return __builtin_amdgcn_wmma_f32_16x16x32_bf16(false, a, false, b, (short)0, c,
                                                 false, false);
}

// ---------------- GEMM: Y = X @ W^T + bias ----------------
// X: [Mtot,K] (float or bf16, row-major), W: [Ntot,K] fp32 row-major.
// Block: 128 thr = 4 waves; wave tile 32(M)x64(N) -> block tile 32x256.
// Software-pipelined k-loop: next step's fragments load behind current WMMAs.
template <typename AT, bool F32OUT>
__global__ __launch_bounds__(128) void gemm_xwt_kernel(
    const AT* __restrict__ X, const float* __restrict__ W,
    const float* __restrict__ bias, void* __restrict__ Yv,
    int Mtot, int Ntot, int Kdim)
{
  const int lane = threadIdx.x & 31;
  const int wave = threadIdx.x >> 5;
  const int m0 = blockIdx.x * 32;
  const int n0 = blockIdx.y * 256 + wave * 64;
  const int r  = lane & 15;
  const int hh = lane >> 4;

  const AT*    a0 = X + (size_t)(m0 + r) * Kdim;
  const AT*    a1 = a0 + (size_t)16 * Kdim;
  const float* w0 = W + (size_t)(n0 + r) * Kdim;

  f32x8 acc0[4] = {}, acc1[4] = {};

  // pipeline prologue: fragments for k = 0
  bf16x16 cA0 = load_afrag(a0, 0, hh);
  bf16x16 cA1 = load_afrag(a1, 0, hh);
  bf16x16 cB[4];
#pragma unroll
  for (int t = 0; t < 4; ++t) cB[t] = load_bfrag(w0 + (size_t)(16 * t) * Kdim, 0, hh);

  for (int k = 32; k < Kdim; k += 32) {
    // issue next step's loads first (overlap with WMMAs below)
    bf16x16 nA0 = load_afrag(a0, k, hh);
    bf16x16 nA1 = load_afrag(a1, k, hh);
    bf16x16 nB[4];
#pragma unroll
    for (int t = 0; t < 4; ++t) nB[t] = load_bfrag(w0 + (size_t)(16 * t) * Kdim, k, hh);
    if (k + 224 < Kdim) {               // ~7 k-steps ahead -> global_prefetch_b8
      __builtin_prefetch(a0 + k + 224, 0, 3);
      __builtin_prefetch(w0 + k + 224, 0, 3);
    }
#pragma unroll
    for (int t = 0; t < 4; ++t) {
      acc0[t] = wmma_bf16(cA0, cB[t], acc0[t]);
      acc1[t] = wmma_bf16(cA1, cB[t], acc1[t]);
    }
    cA0 = nA0; cA1 = nA1;
#pragma unroll
    for (int t = 0; t < 4; ++t) cB[t] = nB[t];
  }
#pragma unroll
  for (int t = 0; t < 4; ++t) {         // pipeline epilogue: last k-step
    acc0[t] = wmma_bf16(cA0, cB[t], acc0[t]);
    acc1[t] = wmma_bf16(cA1, cB[t], acc1[t]);
  }

#pragma unroll
  for (int t = 0; t < 4; ++t) {
    const int n = n0 + 16 * t + r;
    const float bv = bias[n];
#pragma unroll
    for (int i = 0; i < 8; ++i) {
      const int m = m0 + i + 8 * hh;   // D layout: VGPR i -> M = i + 8*(lane>>4)
      const float v0 = acc0[t][i] + bv;
      const float v1 = acc1[t][i] + bv;
      if constexpr (F32OUT) {
        float* Y = (float*)Yv;
        Y[(size_t)m * Ntot + n]        = v0;
        Y[(size_t)(m + 16) * Ntot + n] = v1;
      } else {
        bf16* Y = (bf16*)Yv;
        Y[(size_t)m * Ntot + n]        = (bf16)v0;
        Y[(size_t)(m + 16) * Ntot + n] = (bf16)v1;
      }
    }
  }
}

// ---------------- per-head transpose: kp[B,S,D] -> vT[B*H, DH, S] ----------------
__global__ __launch_bounds__(256) void transpose_heads_kernel(
    const bf16* __restrict__ kp, bf16* __restrict__ vT)
{
  __shared__ bf16 tile[32][33];
  const int bh = blockIdx.z, b = bh >> 4, hd = bh & 15;
  const int d0 = blockIdx.x * 32;
  const int s0 = blockIdx.y * 32;
  const int tx = threadIdx.x, ty = threadIdx.y;   // (32,8)
#pragma unroll
  for (int i = 0; i < 4; ++i) {
    const int s = s0 + ty + 8 * i;
    tile[ty + 8 * i][tx] = kp[((size_t)(b * SQ + s)) * DM + hd * DH + d0 + tx];
  }
  __syncthreads();
#pragma unroll
  for (int i = 0; i < 4; ++i) {
    const int d = d0 + ty + 8 * i;
    vT[((size_t)bh * DH + d) * SQ + s0 + tx] = tile[tx][ty + 8 * i];
  }
}

// ---------------- scores = (q @ k^T) * scale -> fp32 attn region ----------------
// All 8 A fragments preloaded (reused across N tiles); B double-buffered across
// the 4 unrolled k-steps so loads stay in flight behind the WMMAs.
__global__ __launch_bounds__(128) void attn_scores_kernel(
    const bf16* __restrict__ qp, const bf16* __restrict__ kp,
    float* __restrict__ attn)
{
  const int lane = threadIdx.x & 31, wave = threadIdx.x >> 5;
  const int bh = blockIdx.z, b = bh >> 4, hd = bh & 15;
  const int m0 = blockIdx.x * 32;
  const int n0 = blockIdx.y * 256 + wave * 64;
  const int r = lane & 15, hh = lane >> 4;

  const bf16* qb = qp + (size_t)(b * SQ + m0 + r) * DM + hd * DH;
  const bf16* kb = kp + (size_t)(b * SQ + n0 + r) * DM + hd * DH;

  f32x8 acc0[4] = {}, acc1[4] = {};

  bf16x16 A0[4], A1[4];
#pragma unroll
  for (int kk = 0; kk < 4; ++kk) {
    A0[kk] = load_afrag(qb, kk * 32, hh);
    A1[kk] = load_afrag(qb + (size_t)16 * DM, kk * 32, hh);
  }
  bf16x16 cB[4];
#pragma unroll
  for (int t = 0; t < 4; ++t) cB[t] = load_bfrag(kb + (size_t)(16 * t) * DM, 0, hh);

#pragma unroll
  for (int kk = 0; kk < 4; ++kk) {
    bf16x16 nB[4];
    if (kk < 3) {
#pragma unroll
      for (int t = 0; t < 4; ++t)
        nB[t] = load_bfrag(kb + (size_t)(16 * t) * DM, (kk + 1) * 32, hh);
    }
#pragma unroll
    for (int t = 0; t < 4; ++t) {
      acc0[t] = wmma_bf16(A0[kk], cB[t], acc0[t]);
      acc1[t] = wmma_bf16(A1[kk], cB[t], acc1[t]);
    }
    if (kk < 3) {
#pragma unroll
      for (int t = 0; t < 4; ++t) cB[t] = nB[t];
    }
  }

  const float scale = 0.08838834764831845f;  // 1/sqrt(128)
  float* ob = attn + (size_t)bh * SQ * SQ;
#pragma unroll
  for (int t = 0; t < 4; ++t) {
    const int n = n0 + 16 * t + r;
#pragma unroll
    for (int i = 0; i < 8; ++i) {
      const int m = m0 + i + 8 * hh;
      ob[(size_t)m * SQ + n]        = acc0[t][i] * scale;
      ob[(size_t)(m + 16) * SQ + n] = acc1[t][i] * scale;
    }
  }
}

// ---------------- row softmax in place (one block per row of 2048) ----------------
__global__ __launch_bounds__(256) void softmax_kernel(float* __restrict__ attn)
{
  __shared__ float red[256];
  const int tid = threadIdx.x;
  float* row = attn + (size_t)blockIdx.x * SQ;
  float v[8];
  float mx = -3.4e38f;
#pragma unroll
  for (int i = 0; i < 8; ++i) { v[i] = row[tid + 256 * i]; mx = fmaxf(mx, v[i]); }
  red[tid] = mx; __syncthreads();
  for (int s = 128; s > 0; s >>= 1) {
    if (tid < s) red[tid] = fmaxf(red[tid], red[tid + s]);
    __syncthreads();
  }
  mx = red[0]; __syncthreads();
  float sum = 0.f;
#pragma unroll
  for (int i = 0; i < 8; ++i) { v[i] = __expf(v[i] - mx); sum += v[i]; }
  red[tid] = sum; __syncthreads();
  for (int s = 128; s > 0; s >>= 1) {
    if (tid < s) red[tid] += red[tid + s];
    __syncthreads();
  }
  const float inv = 1.f / red[0];
#pragma unroll
  for (int i = 0; i < 8; ++i) row[tid + 256 * i] = v[i] * inv;
}

// ---------------- ctx = attn @ v  (v = k_proj), output bf16 [B,S,D] ----------------
// Block: 4 waves, 2x2 over M(64) x N(128); per-head; K = SQ. Pipelined k-loop.
__global__ __launch_bounds__(128) void attn_ctx_kernel(
    const float* __restrict__ attn, const bf16* __restrict__ vT,
    bf16* __restrict__ cb)
{
  const int lane = threadIdx.x & 31, wave = threadIdx.x >> 5;
  const int wm = wave >> 1, wn = wave & 1;
  const int bh = blockIdx.z, b = bh >> 4, hd = bh & 15;
  const int m0 = blockIdx.x * 64 + wm * 32;
  const int n0 = wn * 64;
  const int r = lane & 15, hh = lane >> 4;

  const float* ar = attn + ((size_t)bh * SQ + (m0 + r)) * SQ;
  const bf16*  vr = vT + ((size_t)bh * DH + (n0 + r)) * SQ;

  f32x8 acc0[4] = {}, acc1[4] = {};

  bf16x16 cA0 = load_afrag(ar, 0, hh);
  bf16x16 cA1 = load_afrag(ar + (size_t)16 * SQ, 0, hh);
  bf16x16 cB[4];
#pragma unroll
  for (int t = 0; t < 4; ++t) cB[t] = load_bfrag(vr + (size_t)(16 * t) * SQ, 0, hh);

  for (int k = 32; k < SQ; k += 32) {
    bf16x16 nA0 = load_afrag(ar, k, hh);
    bf16x16 nA1 = load_afrag(ar + (size_t)16 * SQ, k, hh);
    bf16x16 nB[4];
#pragma unroll
    for (int t = 0; t < 4; ++t) nB[t] = load_bfrag(vr + (size_t)(16 * t) * SQ, k, hh);
    if (k + 224 < SQ) {
      __builtin_prefetch(ar + k + 224, 0, 3);
      __builtin_prefetch(vr + k + 224, 0, 3);
    }
#pragma unroll
    for (int t = 0; t < 4; ++t) {
      acc0[t] = wmma_bf16(cA0, cB[t], acc0[t]);
      acc1[t] = wmma_bf16(cA1, cB[t], acc1[t]);
    }
    cA0 = nA0; cA1 = nA1;
#pragma unroll
    for (int t = 0; t < 4; ++t) cB[t] = nB[t];
  }
#pragma unroll
  for (int t = 0; t < 4; ++t) {
    acc0[t] = wmma_bf16(cA0, cB[t], acc0[t]);
    acc1[t] = wmma_bf16(cA1, cB[t], acc1[t]);
  }

#pragma unroll
  for (int t = 0; t < 4; ++t) {
    const int n = n0 + 16 * t + r;
#pragma unroll
    for (int i = 0; i < 8; ++i) {
      const int m = m0 + i + 8 * hh;
      cb[(size_t)(b * SQ + m) * DM + hd * DH + n]        = (bf16)acc0[t][i];
      cb[(size_t)(b * SQ + m + 16) * DM + hd * DH + n]   = (bf16)acc1[t][i];
    }
  }
}

// ---------------- host-side launcher ----------------
extern "C" void kernel_launch(void* const* d_in, const int* in_sizes, int n_in,
                              void* d_out, int out_size, void* d_ws, size_t ws_size,
                              hipStream_t stream) {
  (void)in_sizes; (void)n_in; (void)out_size; (void)ws_size;
  const float* Q  = (const float*)d_in[0];
  const float* K  = (const float*)d_in[1];
  // d_in[2] = V : dead per reference bug (v = projected k)
  const float* Wq = (const float*)d_in[3];
  const float* bq = (const float*)d_in[4];
  const float* Wk = (const float*)d_in[5];
  const float* bk = (const float*)d_in[6];
  // d_in[7]=Wv, d_in[8]=bv : dead per reference bug
  const float* Wo = (const float*)d_in[9];
  const float* bo = (const float*)d_in[10];

  // workspace: 4 bf16 buffers of BS*DM elements each (16.8 MB each, 67 MB total)
  bf16* qp = (bf16*)d_ws;
  bf16* kp = qp + (size_t)BS * DM;
  bf16* vT = kp + (size_t)BS * DM;   // [B*H, DH, SQ]
  bf16* cb = vT + (size_t)BS * DM;

  float* outp  = (float*)d_out;                        // [B*S, D]
  float* attnp = (float*)d_out + (size_t)BS * DM;      // [B*H, S, S]

  // 1-2) q = Q@Wq^T + bq ; k = K@Wk^T + bk   (bf16 outputs)
  dim3 ggrid(BS / 32, DM / 256);
  gemm_xwt_kernel<float, false><<<ggrid, 128, 0, stream>>>(Q, Wq, bq, qp, BS, DM, DM);
  gemm_xwt_kernel<float, false><<<ggrid, 128, 0, stream>>>(K, Wk, bk, kp, BS, DM, DM);

  // 3) per-head transpose k_proj -> vT
  transpose_heads_kernel<<<dim3(DH / 32, SQ / 32, BB * NH), dim3(32, 8), 0, stream>>>(kp, vT);

  // 4) scores -> attn region of d_out (fp32, scaled)
  attn_scores_kernel<<<dim3(SQ / 32, SQ / 256, BB * NH), 128, 0, stream>>>(qp, kp, attnp);

  // 5) softmax in place (writes final attn_w output)
  softmax_kernel<<<dim3(BB * NH * SQ), 256, 0, stream>>>(attnp);

  // 6) ctx = attn @ v -> bf16 [B,S,D]
  attn_ctx_kernel<<<dim3(SQ / 64, 1, BB * NH), 128, 0, stream>>>(attnp, vT, cb);

  // 7) out = ctx @ Wo^T + bo -> fp32 d_out
  gemm_xwt_kernel<bf16, true><<<ggrid, 128, 0, stream>>>(cb, Wo, bo, outp, BS, DM, DM);
}
